// TinyViT_24515673325836
// MI455X (gfx1250) — matrix operand order repaired and checked by reading, not compiled, not run on hardware
//
#include <hip/hip_runtime.h>

// TinyViT CLS-attention, algebraically collapsed:
//   - only query row 0 (CLS) matters for the output
//   - patch embedding is rank-1 -> K/V rows are scalar*vec+vec
//   - q0/k0/v0 come from the cls token with identity RoPE (r=0)
// One wave32 per batch; both heads packed into the K-split of one
// v_wmma_f32_16x16x32_f16 for (a) q.K scores and (b) P.V accumulation.

#define D_MODEL    32
#define HEAD_DIM   16
#define NUM_CLASSES 10
#define NB         64
#define N_TOK      1024
#define S_TOK      1025
#define NGROUPS    65          // ceil(1025/16)
#define S_PAD      (NGROUPS*16) // 1040

typedef __attribute__((ext_vector_type(16))) _Float16 v16h;
typedef __attribute__((ext_vector_type(8)))  float    v8f;

// ws layout (floats): [0:32) wk  [32:64) ck  [64:96) k0
//                     [96:128) wv [128:160) cv [160:192) v0 [192:224) q0
__global__ __launch_bounds__(32) void tinyvit_setup(
    const float* __restrict__ cls, const float* __restrict__ Wp, const float* __restrict__ bp,
    const float* __restrict__ Wq,  const float* __restrict__ bq,
    const float* __restrict__ Wk,  const float* __restrict__ bk,
    const float* __restrict__ Wv,  const float* __restrict__ bv,
    float* __restrict__ ws)
{
    const int d = threadIdx.x;   // 0..31
    float awk = 0.f, ack = 0.f, ak0 = 0.f;
    float awv = 0.f, acv = 0.f, av0 = 0.f, aq0 = 0.f;
    #pragma unroll
    for (int c = 0; c < 32; ++c) {
        const float wk = Wk[c * 32 + d];
        const float wv = Wv[c * 32 + d];
        const float wq = Wq[c * 32 + d];
        awk += Wp[c]  * wk;  ack += bp[c] * wk;  ak0 += cls[c] * wk;
        awv += Wp[c]  * wv;  acv += bp[c] * wv;  av0 += cls[c] * wv;
        aq0 += cls[c] * wq;
    }
    ws[      d] = awk;
    ws[ 32 + d] = ack + bk[d];
    ws[ 64 + d] = ak0 + bk[d];
    ws[ 96 + d] = awv;
    ws[128 + d] = acv + bv[d];
    ws[160 + d] = av0 + bv[d];
    ws[192 + d] = aq0 + bq[d];
}

__global__ __launch_bounds__(32) void tinyvit_attn(
    const float* __restrict__ x,  const float* __restrict__ rg,
    const float* __restrict__ ws,
    const float* __restrict__ Wf, const float* __restrict__ bf,
    float* __restrict__ out)
{
    __shared__ float s0[S_PAD];        // head0 scores -> probs
    __shared__ float s1[S_PAD];        // head1 scores -> probs
    __shared__ float xb[S_PAD];        // cached pixel values
    __shared__ float ob[D_MODEL];      // attention output (CLS row)

    const int lane = threadIdx.x;      // 0..31
    const int b    = blockIdx.x;       // batch
    const int m    = lane & 15;        // key-in-group / B column / A row id
    const int hi   = lane >> 4;        // which 16-wide half of K this lane covers
    const int c0   = hi * 16;          // model-dim components [c0, c0+16)

    // per-lane register copies of the folded vectors
    float wkr[16], ckr[16], k0r[16], wvr[16], cvr[16], v0r[16];
    #pragma unroll
    for (int i = 0; i < 16; ++i) {
        wkr[i] = ws[      c0 + i];
        ckr[i] = ws[ 32 + c0 + i];
        k0r[i] = ws[ 64 + c0 + i];
        wvr[i] = ws[ 96 + c0 + i];
        cvr[i] = ws[128 + c0 + i];
        v0r[i] = ws[160 + c0 + i];
    }

    // ---- A matrix for scores: row0 = 0.25*q0(head0) in K0..15,
    //                           row1 = 0.25*q0(head1) in K16..31, rest 0.
    // 16-bit A 16x32 layout (contiguous): lanes 0-15 rows M=0..15 elems0-15 <-> K0-15;
    // lanes 16-31 same rows, elems0-15 <-> K16-31.
    const float* q0 = ws + 192;
    v16h qA = {};
    if (m == 0 && hi == 0) {
        #pragma unroll
        for (int j = 0; j < 16; ++j) qA[j] = (_Float16)(0.25f * q0[j]);
    } else if (m == 1 && hi == 1) {
        #pragma unroll
        for (int j = 0; j < 16; ++j) qA[j] = (_Float16)(0.25f * q0[16 + j]);
    }

    // ---- Phase 1: scores[s] for all 1025 keys, 16 keys per WMMA
    for (int g = 0; g < NGROUPS; ++g) {
        const int  s     = g * 16 + m;
        const bool valid = (s < S_TOK);
        float xv = 0.f, rx = 0.f, ry = 0.f;
        if (s >= 1 && valid) {
            const int n = s - 1;
            xv = x[b * N_TOK + n];
            rx = rg[(b * N_TOK + n) * 2 + 0];
            ry = rg[(b * N_TOK + n) * 2 + 1];
        }
        float kc[16];
        if (s == 0) {
            #pragma unroll
            for (int i = 0; i < 16; ++i) kc[i] = k0r[i];   // RoPE(r=0) = identity
        } else {
            #pragma unroll
            for (int i = 0; i < 16; ++i) kc[i] = xv * wkr[i] + ckr[i];
            const float fr[4] = {1.f, 0.1f, 0.01f, 0.001f}; // 10000^(-j/4)
            #pragma unroll
            for (int p = 0; p < 8; ++p) {                  // axial 2D RoPE
                const float a  = (p < 4) ? rx * fr[p] : ry * fr[p - 4];
                const float sp = __sinf(a), cp = __cosf(a);
                const float t1 = kc[2 * p], t2 = kc[2 * p + 1];
                kc[2 * p]     = t1 * cp - t2 * sp;
                kc[2 * p + 1] = t1 * sp + t2 * cp;
            }
        }
        // B column N=m: lane(hi=0) supplies K0..15, lane(hi=1) K16..31
        v16h kB;
        #pragma unroll
        for (int i = 0; i < 16; ++i) kB[i] = (_Float16)kc[i];

        v8f sc = {};
        sc = __builtin_amdgcn_wmma_f32_16x16x32_f16(false, qA, false, kB,
                                                    (short)0, sc, false, false);
        if (hi == 0) {                 // D: lanes0-15, VGPR j -> row M=j, col N=lane
            s0[s] = valid ? sc[0] : -3.0e38f;
            s1[s] = valid ? sc[1] : -3.0e38f;
            xb[s] = xv;
        }
    }
    __syncthreads();

    // ---- Phase 2: per-head softmax over 1025 scores (wave32 reductions)
    float mx0 = -3.0e38f, mx1 = -3.0e38f;
    for (int i = lane; i < S_PAD; i += 32) {
        mx0 = fmaxf(mx0, s0[i]);
        mx1 = fmaxf(mx1, s1[i]);
    }
    #pragma unroll
    for (int o = 16; o > 0; o >>= 1) {
        mx0 = fmaxf(mx0, __shfl_xor(mx0, o, 32));
        mx1 = fmaxf(mx1, __shfl_xor(mx1, o, 32));
    }
    float z0 = 0.f, z1 = 0.f;
    __syncthreads();
    for (int i = lane; i < S_PAD; i += 32) {
        const float p0 = __expf(s0[i] - mx0);
        const float p1 = __expf(s1[i] - mx1);
        s0[i] = p0;  s1[i] = p1;       // keep unnormalized probs in LDS
        z0 += p0;    z1 += p1;
    }
    #pragma unroll
    for (int o = 16; o > 0; o >>= 1) {
        z0 += __shfl_xor(z0, o, 32);
        z1 += __shfl_xor(z1, o, 32);
    }
    const float inv0 = 1.f / z0, inv1 = 1.f / z1;
    __syncthreads();

    // ---- Phase 3: out_h = sum_s p_h[s] * v_s, 16 keys per WMMA, f32 acc
    v8f oacc = {};
    for (int g = 0; g < NGROUPS; ++g) {
        const int base = g * 16;
        v16h pA = {};
        if (m == 0 && hi == 0) {       // row0: head0 probs in K0..15
            #pragma unroll
            for (int j = 0; j < 16; ++j) pA[j] = (_Float16)s0[base + j];
        } else if (m == 1 && hi == 1) { // row1: head1 probs in K16..31
            #pragma unroll
            for (int j = 0; j < 16; ++j) pA[j] = (_Float16)s1[base + j];
        }
        v16h vB;
        #pragma unroll
        for (int i = 0; i < 16; ++i) {
            const int key = base + i;
            float v = 0.f;
            if (key == 0) v = ws[160 + c0 + m];
            else if (key < S_TOK) v = xb[key] * ws[96 + c0 + m] + ws[128 + c0 + m];
            vB[i] = (_Float16)v;
        }
        oacc = __builtin_amdgcn_wmma_f32_16x16x32_f16(false, pA, false, vB,
                                                      (short)0, oacc, false, false);
    }
    if (hi == 0) {
        ob[m]      = oacc[0] * inv0;   // head0 dims 0..15
        ob[16 + m] = oacc[1] * inv1;   // head1 dims 16..31
    }
    __syncthreads();

    // ---- Phase 4: logits = out @ Wf + bf
    if (lane < NUM_CLASSES) {
        float lg = bf[lane];
        #pragma unroll
        for (int d = 0; d < D_MODEL; ++d) lg += ob[d] * Wf[d * NUM_CLASSES + lane];
        out[b * NUM_CLASSES + lane] = lg;
    }
}

extern "C" void kernel_launch(void* const* d_in, const int* in_sizes, int n_in,
                              void* d_out, int out_size, void* d_ws, size_t ws_size,
                              hipStream_t stream) {
    const float* x   = (const float*)d_in[0];
    const float* rg  = (const float*)d_in[1];
    const float* cls = (const float*)d_in[2];
    const float* Wp  = (const float*)d_in[3];
    const float* bp  = (const float*)d_in[4];
    const float* Wq  = (const float*)d_in[5];
    const float* bq  = (const float*)d_in[6];
    const float* Wk  = (const float*)d_in[7];
    const float* bk  = (const float*)d_in[8];
    const float* Wv  = (const float*)d_in[9];
    const float* bv  = (const float*)d_in[10];
    const float* Wf  = (const float*)d_in[11];
    const float* bf  = (const float*)d_in[12];
    float* out = (float*)d_out;
    float* ws  = (float*)d_ws;

    tinyvit_setup<<<1, 32, 0, stream>>>(cls, Wp, bp, Wq, bq, Wk, bk, Wv, bv, ws);
    tinyvit_attn<<<NB, 32, 0, stream>>>(x, rg, ws, Wf, bf, out);
}
